// PointPillarClusterFusion_72026601554649
// MI455X (gfx1250) — matrix operand order repaired and checked by reading, not compiled, not run on hardware
//
#include <hip/hip_runtime.h>
#include <hip/hip_bf16.h>

// ---------------------------------------------------------------------------
// Problem constants (match reference)
// ---------------------------------------------------------------------------
#define NPTS  200000
#define CIN   384
#define NB    2
#define MAXC  8192
#define NCELL (52 * 52 * 2)   // 5408 possible (cx,cy,b) cells

// ---------------------------------------------------------------------------
// Workspace layout (bytes)
// ---------------------------------------------------------------------------
#define OFF_W1T  ((size_t)0)                      // 256x384 bf16
#define OFF_W2T  (OFF_W1T + 256*384*2)            // 128x256 bf16
#define OFF_S1T  (OFF_W2T + 128*256*2)            // 64x128 bf16
#define OFF_P1T  (OFF_S1T + 64*128*2)             // 256x128 bf16
#define OFF_P2T  (OFF_P1T + 256*128*2)            // 768x256 bf16
#define OFF_SC1  (OFF_P2T + 768*256*2)            // 256 f32
#define OFF_SH1  (OFF_SC1 + 256*4)
#define OFF_SC2  (OFF_SH1 + 256*4)                // 64 f32
#define OFF_SH2  (OFF_SC2 + 64*4)
#define OFF_OCC  (OFF_SH2 + 64*4)                 // NCELL i32
#define OFF_RANK (OFF_OCC + NCELL*4)              // NCELL i32
#define OFF_CNT  (OFF_RANK + NCELL*4)             // MAXC f32
#define OFF_FEAT (OFF_CNT + MAXC*4)               // MAXC*128 f32
#define OFF_XYZ  (OFF_FEAT + (size_t)MAXC*128*4)  // MAXC*3 f32
#define OFF_SEG  (OFF_XYZ + MAXC*3*4)             // MAXC f32

// ---------------------------------------------------------------------------
// WMMA plumbing (bf16 16x16x32, f32 accumulate)
// ---------------------------------------------------------------------------
typedef __attribute__((ext_vector_type(16))) __bf16 v16bf;
typedef __attribute__((ext_vector_type(8)))  float  v8f;

// 16B-aligned so fragment loads lower to ds_load_b128 / global_load_b128
struct alignas(16) U128 { unsigned int x, y, z, w; };
struct alignas(16) BF16Frag { U128 lo, hi; };

static __device__ __forceinline__ unsigned short f2bf(float f) {
  unsigned int u = __float_as_uint(f);
  u += 0x7FFFu + ((u >> 16) & 1u);          // round-to-nearest-even
  return (unsigned short)(u >> 16);
}
static __device__ __forceinline__ float bf2f(unsigned short h) {
  return __uint_as_float(((unsigned int)h) << 16);
}

// A fragment (16x32 bf16, row-major source): lane l -> row (l&15),
// K chunks {g*8..g*8+7} and {16+g*8..16+g*8+7}, g = lane>>4.
static __device__ __forceinline__ v16bf load_a16x32(const unsigned short* base,
                                                    int rowStride, int lane) {
  int r = lane & 15, g = (lane >> 4) & 1;
  const unsigned short* p = base + r * rowStride + g * 8;
  BF16Frag f;
  f.lo = *(const U128*)(p);
  f.hi = *(const U128*)(p + 16);
  return __builtin_bit_cast(v16bf, f);
}
// B fragment (32x16 bf16) from transposed weights WT[n][k]:
// lane l -> column (l&15), K chunks {g*8..g*8+7} and {16+g*8..16+g*8+7}.
static __device__ __forceinline__ v16bf load_b32x16(const unsigned short* baseT,
                                                    int rowStride, int lane) {
  int n = lane & 15, g = (lane >> 4) & 1;
  const unsigned short* p = baseT + n * rowStride + g * 8;
  BF16Frag f;
  f.lo = *(const U128*)(p);
  f.hi = *(const U128*)(p + 16);
  return __builtin_bit_cast(v16bf, f);
}
static __device__ __forceinline__ v8f wmma_bf16(v16bf a, v16bf b, v8f c) {
  return __builtin_amdgcn_wmma_f32_16x16x32_bf16(false, a, false, b,
                                                 (short)0, c, false, false);
}

// ---------------------------------------------------------------------------
// Kernel 0: weight transpose/convert, BN folding, accumulator zeroing
// ---------------------------------------------------------------------------
__global__ void k_prep(const float* __restrict__ W1, const float* __restrict__ W2,
                       const float* __restrict__ S1, const float* __restrict__ P1,
                       const float* __restrict__ P2,
                       const float* __restrict__ b1, const float* __restrict__ g1,
                       const float* __restrict__ be1, const float* __restrict__ m1,
                       const float* __restrict__ v1,
                       const float* __restrict__ sb1, const float* __restrict__ g2,
                       const float* __restrict__ be2, const float* __restrict__ m2,
                       const float* __restrict__ v2,
                       unsigned char* __restrict__ ws) {
  int tid = blockIdx.x * blockDim.x + threadIdx.x;
  int stride = gridDim.x * blockDim.x;
  unsigned short* w1t = (unsigned short*)(ws + OFF_W1T);
  unsigned short* w2t = (unsigned short*)(ws + OFF_W2T);
  unsigned short* s1t = (unsigned short*)(ws + OFF_S1T);
  unsigned short* p1t = (unsigned short*)(ws + OFF_P1T);
  unsigned short* p2t = (unsigned short*)(ws + OFF_P2T);
  float* sc1 = (float*)(ws + OFF_SC1);
  float* sh1 = (float*)(ws + OFF_SH1);
  float* sc2 = (float*)(ws + OFF_SC2);
  float* sh2 = (float*)(ws + OFF_SH2);
  int*   occ = (int*)(ws + OFF_OCC);
  int*   rnk = (int*)(ws + OFF_RANK);
  float* cnt = (float*)(ws + OFF_CNT);
  float* fsum = (float*)(ws + OFF_FEAT);
  float* xsum = (float*)(ws + OFF_XYZ);
  float* ssum = (float*)(ws + OFF_SEG);

  for (int i = tid; i < 256 * 384; i += stride) { int n = i / 384, k = i - n * 384; w1t[i] = f2bf(W1[k * 256 + n]); }
  for (int i = tid; i < 128 * 256; i += stride) { int n = i >> 8, k = i & 255; w2t[i] = f2bf(W2[k * 128 + n]); }
  for (int i = tid; i < 64 * 128;  i += stride) { int n = i >> 7, k = i & 127; s1t[i] = f2bf(S1[k * 64 + n]); }
  for (int i = tid; i < 256 * 128; i += stride) { int n = i >> 7, k = i & 127; p1t[i] = f2bf(P1[k * 256 + n]); }
  for (int i = tid; i < 768 * 256; i += stride) { int n = i >> 8, k = i & 255; p2t[i] = f2bf(P2[k * 768 + n]); }
  for (int i = tid; i < 256; i += stride) {
    float s = g1[i] * rsqrtf(v1[i] + 1e-5f);
    sc1[i] = s; sh1[i] = (b1[i] - m1[i]) * s + be1[i];
  }
  for (int i = tid; i < 64; i += stride) {
    float s = g2[i] * rsqrtf(v2[i] + 1e-5f);
    sc2[i] = s; sh2[i] = (sb1[i] - m2[i]) * s + be2[i];
  }
  for (int i = tid; i < NCELL; i += stride) { occ[i] = 0; rnk[i] = 0; }
  for (int i = tid; i < MAXC; i += stride) { cnt[i] = 0.f; ssum[i] = 0.f; }
  for (int i = tid; i < MAXC * 3; i += stride) xsum[i] = 0.f;
  for (size_t i = tid; i < (size_t)MAXC * 128; i += stride) fsum[i] = 0.f;
}

// ---------------------------------------------------------------------------
// Kernel 1: mark occupied cells
// ---------------------------------------------------------------------------
__global__ void k_mark(const float* __restrict__ points, const int* __restrict__ bidx,
                       int* __restrict__ occ) {
  int i = blockIdx.x * blockDim.x + threadIdx.x;
  if (i >= NPTS) return;
  float px = points[i * 4 + 0], py = points[i * 4 + 1];
  int cx = (int)floorf((px + 51.2f) * 0.5f); cx = cx < 0 ? 0 : (cx > 51 ? 51 : cx);
  int cy = (int)floorf((py + 51.2f) * 0.5f); cy = cy < 0 ? 0 : (cy > 51 ? 51 : cy);
  occ[(cx * 52 + cy) * 2 + bidx[i]] = 1;     // benign racing stores of 1
}

// ---------------------------------------------------------------------------
// Kernel 2: exclusive prefix scan over occupancy -> rank (== jnp.unique inverse)
// ---------------------------------------------------------------------------
__global__ void k_scan(const int* __restrict__ occ, int* __restrict__ rnk) {
  __shared__ int partial[256];
  const int CH = 22;                          // 256*22 = 5632 >= NCELL
  int t = threadIdx.x;
  int base = t * CH;
  int s = 0;
  for (int i = 0; i < CH; ++i) { int c = base + i; if (c < NCELL) s += occ[c]; }
  partial[t] = s;
  __syncthreads();
  for (int off = 1; off < 256; off <<= 1) {
    int v = partial[t];
    int add = (t >= off) ? partial[t - off] : 0;
    __syncthreads();
    partial[t] = v + add;
    __syncthreads();
  }
  int run = (t == 0) ? 0 : partial[t - 1];
  for (int i = 0; i < CH; ++i) {
    int c = base + i;
    if (c < NCELL) { rnk[c] = run; run += occ[c]; }
  }
}

// ---------------------------------------------------------------------------
// Kernel 3: point pipeline — gather, 3 WMMA layers, seg head, cluster atomics
// Block = 256 threads (8 waves), 128 points/block (16 rows per wave).
// ---------------------------------------------------------------------------
__global__ void __launch_bounds__(256)
k_points(const float* __restrict__ points, const int* __restrict__ bidx,
         const float* __restrict__ bev,
         const float* __restrict__ b2, const float* __restrict__ S2,
         const float* __restrict__ sb2,
         unsigned char* __restrict__ ws,
         float* __restrict__ outPF, float* __restrict__ outSeg) {
  extern __shared__ unsigned char smem[];
  unsigned short* pfLds  = (unsigned short*)smem;          // 128*384 bf16 = 96KB
  unsigned short* hLds   = pfLds + 128 * 384;              // 128*256 bf16 = 64KB
  unsigned short* pf2Lds = hLds + 128 * 256;               // 128*128 bf16 = 32KB
  unsigned short* h2Lds  = pf2Lds + 128 * 128;             // 128*64  bf16 = 16KB
  int*   baseOffLds = (int*)(h2Lds + 128 * 64);
  int*   invLds     = baseOffLds + 128;
  float* probLds    = (float*)(invLds + 128);

  const unsigned short* w1t = (const unsigned short*)(ws + OFF_W1T);
  const unsigned short* w2t = (const unsigned short*)(ws + OFF_W2T);
  const unsigned short* s1t = (const unsigned short*)(ws + OFF_S1T);
  const float* sc1 = (const float*)(ws + OFF_SC1);
  const float* sh1 = (const float*)(ws + OFF_SH1);
  const float* sc2 = (const float*)(ws + OFF_SC2);
  const float* sh2 = (const float*)(ws + OFF_SH2);
  const int*   rnk = (const int*)(ws + OFF_RANK);
  float* cnt  = (float*)(ws + OFF_CNT);
  float* fsum = (float*)(ws + OFF_FEAT);
  float* xsum = (float*)(ws + OFF_XYZ);
  float* ssum = (float*)(ws + OFF_SEG);

  int tid = threadIdx.x;
  int blockBase = blockIdx.x * 128;

  // --- per-point metadata ---
  if (tid < 128) {
    int p = blockBase + tid;
    int pc = p < NPTS ? p : NPTS - 1;
    float px = points[pc * 4 + 0], py = points[pc * 4 + 1];
    int gx = (int)floorf((px + 51.2f) * 2.5f); gx = gx < 0 ? 0 : (gx > 255 ? 255 : gx);
    int gy = (int)floorf((py + 51.2f) * 2.5f); gy = gy < 0 ? 0 : (gy > 255 ? 255 : gy);
    int b = bidx[pc];
    baseOffLds[tid] = (b * CIN) * 65536 + gy * 256 + gx;
    int cx = (int)floorf((px + 51.2f) * 0.5f); cx = cx < 0 ? 0 : (cx > 51 ? 51 : cx);
    int cy = (int)floorf((py + 51.2f) * 0.5f); cy = cy < 0 ? 0 : (cy > 51 ? 51 : cy);
    invLds[tid] = rnk[(cx * 52 + cy) * 2 + b];
  }
  __syncthreads();

  // --- gather BEV features -> bf16 LDS (scattered 4B reads, HBM-bound) ---
  for (int idx = tid; idx < 128 * CIN; idx += 256) {
    int p = idx / CIN;
    int c = idx - p * CIN;
    float v = 0.f;
    if (blockBase + p < NPTS) v = bev[(size_t)(baseOffLds[p] + c * 65536)];
    pfLds[p * CIN + c] = f2bf(v);
  }
  __syncthreads();

  int wave = tid >> 5, lane = tid & 31;
  int g = (lane >> 4) & 1, nl = lane & 15;

  // --- layer 1: pf @ W1, folded BN, relu -> hLds ---
  const unsigned short* A1 = pfLds + wave * 16 * CIN;
  for (int n0 = 0; n0 < 256; n0 += 16) {
    v8f acc = {};
    for (int k0 = 0; k0 < CIN; k0 += 32)
      acc = wmma_bf16(load_a16x32(A1 + k0, CIN, lane),
                      load_b32x16(w1t + n0 * CIN + k0, CIN, lane), acc);
    int col = n0 + nl;
    float s = sc1[col], sh = sh1[col];
#pragma unroll
    for (int e = 0; e < 8; ++e) {
      float v = acc[e] * s + sh;
      v = v > 0.f ? v : 0.f;
      hLds[(wave * 16 + e + 8 * g) * 256 + col] = f2bf(v);
    }
  }

  // --- layer 2: h @ W2 + b2 -> point_feats (global f32 + LDS bf16) ---
  const unsigned short* A2 = hLds + wave * 16 * 256;
  for (int n0 = 0; n0 < 128; n0 += 16) {
    v8f acc = {};
    for (int k0 = 0; k0 < 256; k0 += 32)
      acc = wmma_bf16(load_a16x32(A2 + k0, 256, lane),
                      load_b32x16(w2t + n0 * 256 + k0, 256, lane), acc);
    int col = n0 + nl;
    float bias = b2[col];
#pragma unroll
    for (int e = 0; e < 8; ++e) {
      int pl = wave * 16 + e + 8 * g;
      float v = acc[e] + bias;
      pf2Lds[pl * 128 + col] = f2bf(v);
      int p = blockBase + pl;
      if (p < NPTS) outPF[(size_t)p * 128 + col] = v;
    }
  }

  // --- layer 3: pf2 @ S1, folded BN, relu -> h2Lds ---
  const unsigned short* A3 = pf2Lds + wave * 16 * 128;
  for (int n0 = 0; n0 < 64; n0 += 16) {
    v8f acc = {};
    for (int k0 = 0; k0 < 128; k0 += 32)
      acc = wmma_bf16(load_a16x32(A3 + k0, 128, lane),
                      load_b32x16(s1t + n0 * 128 + k0, 128, lane), acc);
    int col = n0 + nl;
    float s = sc2[col], sh = sh2[col];
#pragma unroll
    for (int e = 0; e < 8; ++e) {
      float v = acc[e] * s + sh;
      v = v > 0.f ? v : 0.f;
      h2Lds[(wave * 16 + e + 8 * g) * 64 + col] = f2bf(v);
    }
  }

  // --- seg head: one point per lane (lanes 0..15) ---
  if (lane < 16) {
    int pl = wave * 16 + lane;
    float s = sb2[0];
    const unsigned short* hr = h2Lds + pl * 64;
    for (int j = 0; j < 64; ++j) s += bf2f(hr[j]) * S2[j];
    int p = blockBase + pl;
    if (p < NPTS) outSeg[p] = s;
    probLds[pl] = 1.f / (1.f + expf(-s));
  }

  // --- cluster accumulation (fp32 atomics into workspace) ---
  for (int pi = 0; pi < 16; ++pi) {
    int pl = wave * 16 + pi;
    int p = blockBase + pl;
    if (p >= NPTS) break;                     // uniform across wave
    int iv = invLds[pl];
    const float* row = outPF + (size_t)p * 128;
    for (int j = lane; j < 128; j += 32)
      atomicAdd(&fsum[(size_t)iv * 128 + j], row[j]);
    if (lane < 3) atomicAdd(&xsum[iv * 3 + lane], points[p * 4 + lane]);
    if (lane == 0) { atomicAdd(&cnt[iv], 1.0f); atomicAdd(&ssum[iv], probLds[pl]); }
  }
}

// ---------------------------------------------------------------------------
// Kernel 4: cluster MLP (8192x128 -> 256 -> 768) + xyz/seg means
// Block = 256 threads (8 waves), 128 cluster rows/block.
// ---------------------------------------------------------------------------
__global__ void __launch_bounds__(256)
k_cluster(unsigned char* __restrict__ ws,
          const float* __restrict__ pb1, const float* __restrict__ pb2,
          float* __restrict__ outProj, float* __restrict__ outXyz,
          float* __restrict__ outSegC) {
  extern __shared__ unsigned char smem[];
  unsigned short* cfLds = (unsigned short*)smem;           // 128*128 bf16 = 32KB
  unsigned short* t1Lds = cfLds + 128 * 128;               // 128*256 bf16 = 64KB

  const unsigned short* p1t = (const unsigned short*)(ws + OFF_P1T);
  const unsigned short* p2t = (const unsigned short*)(ws + OFF_P2T);
  const float* cnt  = (const float*)(ws + OFF_CNT);
  const float* fsum = (const float*)(ws + OFF_FEAT);
  const float* xsum = (const float*)(ws + OFF_XYZ);
  const float* ssum = (const float*)(ws + OFF_SEG);

  int tid = threadIdx.x;
  int rowBase = blockIdx.x * 128;

  for (int idx = tid; idx < 128 * 128; idx += 256) {
    int r = idx >> 7, j = idx & 127;
    int row = rowBase + r;
    float d = cnt[row]; d = d > 1.f ? d : 1.f;
    cfLds[idx] = f2bf(fsum[(size_t)row * 128 + j] / d);
  }
  for (int idx = tid; idx < 128 * 4; idx += 256) {
    int r = idx >> 2, c = idx & 3;
    int row = rowBase + r;
    float d = cnt[row]; d = d > 1.f ? d : 1.f;
    if (c < 3) outXyz[row * 3 + c] = xsum[row * 3 + c] / d;
    else       outSegC[row] = ssum[row] / d;
  }
  __syncthreads();

  int wave = tid >> 5, lane = tid & 31;
  int g = (lane >> 4) & 1, nl = lane & 15;

  const unsigned short* A1 = cfLds + wave * 16 * 128;
  for (int n0 = 0; n0 < 256; n0 += 16) {
    v8f acc = {};
    for (int k0 = 0; k0 < 128; k0 += 32)
      acc = wmma_bf16(load_a16x32(A1 + k0, 128, lane),
                      load_b32x16(p1t + n0 * 128 + k0, 128, lane), acc);
    int col = n0 + nl;
    float bias = pb1[col];
#pragma unroll
    for (int e = 0; e < 8; ++e) {
      float v = acc[e] + bias;
      v = v > 0.f ? v : 0.f;
      t1Lds[(wave * 16 + e + 8 * g) * 256 + col] = f2bf(v);
    }
  }

  const unsigned short* A2 = t1Lds + wave * 16 * 256;
  for (int n0 = 0; n0 < 768; n0 += 16) {
    v8f acc = {};
    for (int k0 = 0; k0 < 256; k0 += 32)
      acc = wmma_bf16(load_a16x32(A2 + k0, 256, lane),
                      load_b32x16(p2t + n0 * 256 + k0, 256, lane), acc);
    int col = n0 + nl;
    float bias = pb2[col];
#pragma unroll
    for (int e = 0; e < 8; ++e)
      outProj[(size_t)(rowBase + wave * 16 + e + 8 * g) * 768 + col] = acc[e] + bias;
  }
}

// ---------------------------------------------------------------------------
// Host entry
// ---------------------------------------------------------------------------
extern "C" void kernel_launch(void* const* d_in, const int* in_sizes, int n_in,
                              void* d_out, int out_size, void* d_ws, size_t ws_size,
                              hipStream_t stream) {
  const float* points = (const float*)d_in[0];
  const int*   bidx   = (const int*)d_in[1];
  const float* bev    = (const float*)d_in[2];
  const float* W1  = (const float*)d_in[3];
  const float* b1  = (const float*)d_in[4];
  const float* g1  = (const float*)d_in[5];
  const float* be1 = (const float*)d_in[6];
  const float* m1  = (const float*)d_in[7];
  const float* v1  = (const float*)d_in[8];
  const float* W2  = (const float*)d_in[9];
  const float* b2  = (const float*)d_in[10];
  const float* S1  = (const float*)d_in[11];
  const float* sb1 = (const float*)d_in[12];
  const float* g2  = (const float*)d_in[13];
  const float* be2 = (const float*)d_in[14];
  const float* m2  = (const float*)d_in[15];
  const float* v2  = (const float*)d_in[16];
  const float* S2  = (const float*)d_in[17];
  const float* sb2 = (const float*)d_in[18];
  const float* P1  = (const float*)d_in[19];
  const float* pb1 = (const float*)d_in[20];
  const float* P2  = (const float*)d_in[21];
  const float* pb2 = (const float*)d_in[22];
  unsigned char* ws = (unsigned char*)d_ws;

  float* out = (float*)d_out;
  float* outPF   = out;                                   // N x 128
  float* outSeg  = outPF + (size_t)NPTS * 128;            // N x 1
  float* outProj = outSeg + NPTS;                         // MAXC x 768
  float* outXyz  = outProj + (size_t)MAXC * 768;          // MAXC x 3
  float* outSegC = outXyz + (size_t)MAXC * 3;             // MAXC x 1

  k_prep<<<512, 256, 0, stream>>>(W1, W2, S1, P1, P2, b1, g1, be1, m1, v1,
                                  sb1, g2, be2, m2, v2, ws);
  k_mark<<<(NPTS + 255) / 256, 256, 0, stream>>>(points, bidx, (int*)(ws + OFF_OCC));
  k_scan<<<1, 256, 0, stream>>>((const int*)(ws + OFF_OCC), (int*)(ws + OFF_RANK));

  // dynamic LDS: 96K + 64K + 32K + 16K + 3*512B metadata = 214528 B (< 320KB/WGP)
  k_points<<<(NPTS + 127) / 128, 256, 214528, stream>>>(points, bidx, bev,
                                                        b2, S2, sb2, ws, outPF, outSeg);
  // dynamic LDS: 32K + 64K = 98304 B
  k_cluster<<<MAXC / 128, 256, 98304, stream>>>(ws, pb1, pb2, outProj, outXyz, outSegC);
}
